// ResidualVectorQuantizer_7645041787348
// MI455X (gfx1250) — matrix operand order repaired and checked by reading, not compiled, not run on hardware
//
#include <hip/hip_runtime.h>
#include <math.h>

#define RVQ_N      131072
#define RVQ_D      128
#define RVQ_K      1024
#define RVQ_L      4
#define RVQ_BETA   0.25f
#define RVQ_LOGK   6.9314718055994531f   /* ln(1024) */

typedef __attribute__((ext_vector_type(16))) __bf16 v16bf;
typedef __attribute__((ext_vector_type(8)))  float  v8f;

union Frag { v16bf v; uint4 q[2]; };

__device__ __forceinline__ unsigned short f2bf_rn(float f) {
  unsigned u = __float_as_uint(f);
  unsigned r = u + 0x7FFFu + ((u >> 16) & 1u);   // round-to-nearest-even
  return (unsigned short)(r >> 16);
}
__device__ __forceinline__ float bf2f(unsigned short h) {
  return __uint_as_float(((unsigned)h) << 16);
}

// ---------------------------------------------------------------------------
// Zero the two scalar-loss output slots (harness poisons d_out).
// ---------------------------------------------------------------------------
__global__ void rvq_zero(float* __restrict__ s) {
  if (threadIdx.x < 2) s[threadIdx.x] = 0.0f;
}

// ---------------------------------------------------------------------------
// Pre-convert codebooks f32 -> bf16 hi/lo (split precision) + per-code norms.
// ---------------------------------------------------------------------------
__global__ void rvq_prep(const float* __restrict__ cb,
                         unsigned* __restrict__ hi2,
                         unsigned* __restrict__ lo2,
                         float* __restrict__ norms) {
  int c = blockIdx.x * blockDim.x + threadIdx.x;
  if (c >= RVQ_L * RVQ_K) return;
  const float* row = cb + (size_t)c * RVQ_D;
  float ss = 0.0f;
  #pragma unroll 4
  for (int i = 0; i < RVQ_D; i += 2) {
    float a = row[i], b = row[i + 1];
    ss += a * a + b * b;
    unsigned short ha = f2bf_rn(a), hb = f2bf_rn(b);
    unsigned short la = f2bf_rn(a - bf2f(ha)), lb = f2bf_rn(b - bf2f(hb));
    hi2[(size_t)c * (RVQ_D / 2) + (i >> 1)] = (unsigned)ha | ((unsigned)hb << 16);
    lo2[(size_t)c * (RVQ_D / 2) + (i >> 1)] = (unsigned)la | ((unsigned)lb << 16);
  }
  norms[c] = ss;
}

// ---------------------------------------------------------------------------
// Main RVQ kernel. 256 threads = 8 waves; each wave owns 16 rows.
// Per level: split residual to bf16 hi/lo in LDS, WMMA 16x16x32 bf16 over the
// 1024-code codebook in 16-code tiles (3 WMMAs per k-step: hh + lh + hl),
// branchless argmax + softmax stats on s = 2*r.e - |e|^2, then gather/update.
// ---------------------------------------------------------------------------
__global__ __launch_bounds__(256, 1)
void rvq_main(const float* __restrict__ x,
              const float* __restrict__ cb,
              const unsigned short* __restrict__ cb_hi,
              const unsigned short* __restrict__ cb_lo,
              const float* __restrict__ cnorm,
              float* __restrict__ out,
              float* __restrict__ codes,
              float* __restrict__ scalars) {
  __shared__ __align__(16) unsigned short sh_hi[8 * 16 * RVQ_D];   // 32 KB
  __shared__ __align__(16) unsigned short sh_lo[8 * 16 * RVQ_D];   // 32 KB
  __shared__ int sh_idx[8][16];

  const int lane = threadIdx.x & 31;
  const int wv   = threadIdx.x >> 5;
  const int col  = lane & 15;   // 0..15
  const int half = lane >> 4;   // 0 / 1
  const int rowbase = blockIdx.x * 128 + wv * 16;

  // This lane owns residual of row (rowbase+col), dims [half*64, half*64+64).
  const int myrow = rowbase + col;
  const int dimb  = half * 64;
  const float* xrow = x + (size_t)myrow * RVQ_D + dimb;

  float r[64];
  #pragma unroll
  for (int i = 0; i < 16; i++) {
    float4 v = *(const float4*)(xrow + 4 * i);
    r[4*i+0] = v.x; r[4*i+1] = v.y; r[4*i+2] = v.z; r[4*i+3] = v.w;
  }

  unsigned short* myhi = &sh_hi[(wv * 16 + col) * RVQ_D + dimb];
  unsigned short* mylo = &sh_lo[(wv * 16 + col) * RVQ_D + dimb];

  float commit_acc = 0.0f, usage_acc = 0.0f;

  for (int l = 0; l < RVQ_L; l++) {
    // ---- residual -> bf16 hi/lo into LDS ----
    #pragma unroll
    for (int i = 0; i < 64; i += 2) {
      unsigned short h0 = f2bf_rn(r[i]);
      unsigned short h1 = f2bf_rn(r[i + 1]);
      unsigned short g0 = f2bf_rn(r[i]     - bf2f(h0));
      unsigned short g1 = f2bf_rn(r[i + 1] - bf2f(h1));
      *(unsigned*)&myhi[i] = (unsigned)h0 | ((unsigned)h1 << 16);
      *(unsigned*)&mylo[i] = (unsigned)g0 | ((unsigned)g1 << 16);
    }
    __syncthreads();

    // ---- A fragments: 16-bit A 16x32 layout (two 16B chunks per k-step) ----
    Frag ahi[4], alo[4];
    const unsigned short* arow_hi = &sh_hi[(wv * 16 + col) * RVQ_D];
    const unsigned short* arow_lo = &sh_lo[(wv * 16 + col) * RVQ_D];
    #pragma unroll
    for (int s = 0; s < 4; s++) {
      int k0 = 32 * s + 16 * half;
      ahi[s].q[0] = *(const uint4*)(arow_hi + k0);
      ahi[s].q[1] = *(const uint4*)(arow_hi + k0 + 8);
      alo[s].q[0] = *(const uint4*)(arow_lo + k0);
      alo[s].q[1] = *(const uint4*)(arow_lo + k0 + 8);
    }

    // Per-register-set (row m = q + 8*half) stats over this lane's column
    // subset {col, col+16, ...}: branchless argmax + unshifted softmax sums.
    // (usage term is shift-invariant; scores here are O(0.1), exp is safe.)
    float best[8], bZ[8], bW[8];
    int   bidx[8];
    #pragma unroll
    for (int q = 0; q < 8; q++) {
      best[q] = -1e30f; bZ[q] = 0.0f; bW[q] = 0.0f; bidx[q] = 0;
    }

    const int cbase = l * RVQ_K;
    for (int t = 0; t < RVQ_K / 16; t++) {
      const int code = t * 16 + col;
      // B fragment: same contiguous k order as A (not the ISA table order); lane-half = 16 consecutive K.
      const unsigned short* bh = cb_hi + (size_t)(cbase + code) * RVQ_D + 16 * half;
      const unsigned short* bl = cb_lo + (size_t)(cbase + code) * RVQ_D + 16 * half;

      // Load all 8 B fragments up front (one long load clause).
      Frag bhf[4], blf[4];
      #pragma unroll
      for (int s = 0; s < 4; s++) {
        bhf[s].q[0] = *(const uint4*)(bh + 32 * s);
        bhf[s].q[1] = *(const uint4*)(bh + 32 * s + 8);
        blf[s].q[0] = *(const uint4*)(bl + 32 * s);
        blf[s].q[1] = *(const uint4*)(bl + 32 * s + 8);
      }
      float cn = cnorm[cbase + code];

      // Two accumulator chains: acc0 = hi*hi, acc1 = lo*hi + hi*lo.
      v8f acc0 = {0.f, 0.f, 0.f, 0.f, 0.f, 0.f, 0.f, 0.f};
      v8f acc1 = {0.f, 0.f, 0.f, 0.f, 0.f, 0.f, 0.f, 0.f};
      #pragma unroll
      for (int s = 0; s < 4; s++) {
        acc0 = __builtin_amdgcn_wmma_f32_16x16x32_bf16(false, ahi[s].v, false, bhf[s].v,
                                                       (short)0, acc0, false, false);
        acc1 = __builtin_amdgcn_wmma_f32_16x16x32_bf16(false, alo[s].v, false, bhf[s].v,
                                                       (short)0, acc1, false, false);
        acc1 = __builtin_amdgcn_wmma_f32_16x16x32_bf16(false, ahi[s].v, false, blf[s].v,
                                                       (short)0, acc1, false, false);
      }

      #pragma unroll
      for (int q = 0; q < 8; q++) {
        float tv = fmaf(2.0f, acc0[q] + acc1[q], -cn);  // = |r|^2 - dist (shifted)
        bool gt  = tv > best[q];
        bidx[q]  = gt ? code : bidx[q];
        best[q]  = gt ? tv : best[q];
        float e  = __expf(fminf(tv, 80.0f));
        bZ[q]   += e;
        bW[q]    = fmaf(tv, e, bW[q]);
      }
    }

    // ---- cross-lane reduce within each 16-lane half ----
    #pragma unroll
    for (int off = 1; off < 16; off <<= 1) {
      #pragma unroll
      for (int q = 0; q < 8; q++) {
        float ob = __shfl_xor(best[q], off, 32);
        int   oi = __shfl_xor(bidx[q], off, 32);
        bool take = (ob > best[q]) || (ob == best[q] && oi < bidx[q]);
        best[q] = take ? ob : best[q];
        bidx[q] = take ? oi : bidx[q];
        bZ[q] += __shfl_xor(bZ[q], off, 32);
        bW[q] += __shfl_xor(bW[q], off, 32);
      }
    }

    if (col == 0) {
      #pragma unroll
      for (int q = 0; q < 8; q++) {
        int row = half * 8 + q;
        sh_idx[wv][row] = bidx[q];
        // usage row term: logK + E[s] - lse(s)   (shift-invariant form)
        usage_acc += RVQ_LOGK + bW[q] / bZ[q] - __logf(bZ[q]);
        codes[(size_t)(rowbase + row) * RVQ_L + l] = (float)bidx[q];
      }
    }
    __syncthreads();

    // ---- gather quantized (f32 codebook) & update residual ----
    int qidx = sh_idx[wv][col];
    const float* qrow = cb + (size_t)(cbase + qidx) * RVQ_D + dimb;
    #pragma unroll
    for (int i = 0; i < 16; i++) {
      float4 qv = *(const float4*)(qrow + 4 * i);
      r[4*i+0] -= qv.x; r[4*i+1] -= qv.y; r[4*i+2] -= qv.z; r[4*i+3] -= qv.w;
      commit_acc += r[4*i+0]*r[4*i+0] + r[4*i+1]*r[4*i+1]
                  + r[4*i+2]*r[4*i+2] + r[4*i+3]*r[4*i+3];
    }
    __syncthreads();
  }

  // ---- out = quantized_sum = x - final residual ----
  float* orow = out + (size_t)myrow * RVQ_D + dimb;
  #pragma unroll
  for (int i = 0; i < 16; i++) {
    float4 xv = *(const float4*)(xrow + 4 * i);
    float4 ov;
    ov.x = xv.x - r[4*i+0]; ov.y = xv.y - r[4*i+1];
    ov.z = xv.z - r[4*i+2]; ov.w = xv.w - r[4*i+3];
    *(float4*)(orow + 4 * i) = ov;
  }

  // ---- wave-reduce losses, one atomic per wave ----
  #pragma unroll
  for (int off = 1; off < 32; off <<= 1) {
    commit_acc += __shfl_xor(commit_acc, off, 32);
    usage_acc  += __shfl_xor(usage_acc,  off, 32);
  }
  if (lane == 0) {
    atomicAdd(&scalars[0], commit_acc * ((1.0f + RVQ_BETA) / ((float)RVQ_N * (float)RVQ_D)));
    atomicAdd(&scalars[1], usage_acc * (1.0f / (float)RVQ_N));
  }
}

// ---------------------------------------------------------------------------
extern "C" void kernel_launch(void* const* d_in, const int* in_sizes, int n_in,
                              void* d_out, int out_size, void* d_ws, size_t ws_size,
                              hipStream_t stream) {
  (void)in_sizes; (void)n_in; (void)out_size; (void)ws_size;
  const float* x  = (const float*)d_in[0];
  const float* cb = (const float*)d_in[1];

  float* out    = (float*)d_out;
  float* codes  = out + (size_t)RVQ_N * RVQ_D;
  float* scal   = codes + (size_t)RVQ_N * RVQ_L;

  // Workspace: bf16-hi codebook (1 MB) | bf16-lo codebook (1 MB) | norms (16 KB)
  unsigned short* cb_hi = (unsigned short*)d_ws;
  unsigned short* cb_lo = cb_hi + (size_t)RVQ_L * RVQ_K * RVQ_D;
  float* norms = (float*)(cb_lo + (size_t)RVQ_L * RVQ_K * RVQ_D);

  rvq_zero<<<1, 64, 0, stream>>>(scal);
  rvq_prep<<<(RVQ_L * RVQ_K + 255) / 256, 256, 0, stream>>>(
      cb, (unsigned*)cb_hi, (unsigned*)cb_lo, norms);
  rvq_main<<<RVQ_N / 128, 256, 0, stream>>>(
      x, cb, cb_hi, cb_lo, norms, out, codes, scal);
}
